// QuantumEncoder_48490180771952
// MI455X (gfx1250) — matrix operand order repaired and checked by reading, not compiled, not run on hardware
//
#include <hip/hip_runtime.h>
#include <hip/hip_bf16.h>
#include <math.h>

typedef __attribute__((ext_vector_type(16))) _Float16 v16h;
typedef __attribute__((ext_vector_type(8)))  _Float16 v8h;
typedef __attribute__((ext_vector_type(8)))  float    v8f;
typedef __attribute__((ext_vector_type(4)))  float    v4f;

#define QDIM 64

// ---------------------------------------------------------------------------
// Kernel 1: fold the whole circuit into one 64x64 complex matrix
//   M = P*U6 * P*U5 * ... * P*U1 * QFT     (P = CNOT-ring permutation)
// Output: Mout[0..4095]   = M_re (f16, row-major, row = output index)
//         Mout[4096..8191]= M_im (f16, row-major)
// Single workgroup; negligible runtime.
// ---------------------------------------------------------------------------
__global__ __launch_bounds__(64) void build_M_kernel(
    const float* __restrict__ wrot, _Float16* __restrict__ Mout) {
  __shared__ float sMre[QDIM][QDIM];
  __shared__ float sMim[QDIM][QDIM];
  __shared__ float sU[QDIM][QDIM];
  __shared__ int   sperm[QDIM];

  const int t = threadIdx.x;  // 0..63, owns row t

  // CNOT ring composite permutation: forward map m, scatter inverse.
  {
    int m = t;
#pragma unroll
    for (int q = 0; q < 6; ++q) {
      const int cb = 5 - q;
      const int tb = 5 - ((q + 1) % 6);
      if ((m >> cb) & 1) m ^= (1 << tb);
    }
    sperm[m] = t;  // perm = argsort(m): psi_out[i] = psi_in[perm[i]]
  }

  // QFT init: F[j,k] = exp(2*pi*i*j*k/64)/8, use periodicity for accuracy.
  const float w0 = 6.28318530717958647692f / 64.0f;
  for (int k = 0; k < QDIM; ++k) {
    const float ang = w0 * (float)((t * k) & 63);
    sMre[t][k] = cosf(ang) * 0.125f;
    sMim[t][k] = sinf(ang) * 0.125f;
  }
  __syncthreads();

  for (int l = 0; l < 6; ++l) {
    float c[6], s[6];
#pragma unroll
    for (int q = 0; q < 6; ++q) {
      const float th = 0.5f * wrot[l * 6 + q];
      c[q] = cosf(th);
      s[q] = sinf(th);
    }
    // U row t: kron of per-qubit RY gates; wire 0 = MSB.
    for (int j = 0; j < QDIM; ++j) {
      float p = 1.0f;
#pragma unroll
      for (int q = 0; q < 6; ++q) {
        const int ib = (t >> (5 - q)) & 1;
        const int jb = (j >> (5 - q)) & 1;
        p *= (ib == jb) ? c[q] : ((ib == 0) ? -s[q] : s[q]);
      }
      sU[t][j] = p;
    }
    __syncthreads();

    // M_new[t][:] = U[perm[t]][:] @ M   (row-permute fused with the matmul)
    const int pi = sperm[t];
    float nre[QDIM], nim[QDIM];
#pragma unroll
    for (int j = 0; j < QDIM; ++j) { nre[j] = 0.0f; nim[j] = 0.0f; }
    for (int k = 0; k < QDIM; ++k) {
      const float u = sU[pi][k];
#pragma unroll
      for (int j = 0; j < QDIM; ++j) {
        nre[j] = fmaf(u, sMre[k][j], nre[j]);
        nim[j] = fmaf(u, sMim[k][j], nim[j]);
      }
    }
    __syncthreads();
#pragma unroll
    for (int j = 0; j < QDIM; ++j) { sMre[t][j] = nre[j]; sMim[t][j] = nim[j]; }
    __syncthreads();
  }

  for (int k = 0; k < QDIM; ++k) {
    Mout[t * QDIM + k]        = (_Float16)sMre[t][k];
    Mout[4096 + t * QDIM + k] = (_Float16)sMim[t][k];
  }
}

// ---------------------------------------------------------------------------
// Kernel 2: out[b, j] = ((Mre@x_b)_j^2 + (Mim@x_b)_j^2) / ||x_b||^2
// GEMM form: C[16 x 64] = A[16 x 64] * B[64 x 64], A = x tile, B = M^T,
// via v_wmma_f32_16x16x32_f16. One wave per 16-row batch tile; B fragments
// (the folded matrix, 16 KB) are loop-invariant and live in registers.
// ---------------------------------------------------------------------------
__global__ __launch_bounds__(256) void qenc_wmma_kernel(
    const float* __restrict__ x,
    const _Float16* __restrict__ Mre,
    const _Float16* __restrict__ Mim,
    float* __restrict__ out,
    int numTiles) {
  const int lane = threadIdx.x & 31;
  const int wid  = threadIdx.x >> 5;
  const int n    = lane & 15;   // column within 16-wide tile / A row
  const int half = lane >> 4;   // which K half this lane services

  // ---- B fragments: B[k][n] = M[j0+n][k0+k]; lane holds col n, K=16*half+e
  v16h bre[4][2], bim[4][2];
#pragma unroll
  for (int jt = 0; jt < 4; ++jt) {
    const _Float16* rRe = Mre + (jt * 16 + n) * QDIM;
    const _Float16* rIm = Mim + (jt * 16 + n) * QDIM;
#pragma unroll
    for (int kc = 0; kc < 2; ++kc) {
      const int kb = kc * 32 + half * 16;
      v8h rlo = *(const v8h*)(rRe + kb);
      v8h rhi = *(const v8h*)(rRe + kb + 8);
      bre[jt][kc] = __builtin_shufflevector(rlo, rhi, 0, 1, 2, 3, 4, 5, 6, 7,
                                            8, 9, 10, 11, 12, 13, 14, 15);
      v8h ilo = *(const v8h*)(rIm + kb);
      v8h ihi = *(const v8h*)(rIm + kb + 8);
      bim[jt][kc] = __builtin_shufflevector(ilo, ihi, 0, 1, 2, 3, 4, 5, 6, 7,
                                            8, 9, 10, 11, 12, 13, 14, 15);
    }
  }

  const int totalWaves = gridDim.x * 8;
  for (int tile = blockIdx.x * 8 + wid; tile < numTiles; tile += totalWaves) {
    const size_t b0 = (size_t)tile * 16;
    const float* xr = x + (b0 + (size_t)n) * QDIM;  // lane's batch row

    // ---- A fragments (16x32 f16): lane row = n; K = 16*half+0..15.
    //      Convert f32->f16, accumulate ||x||^2.
    float sumsq = 0.0f;
    v16h a[2];
#pragma unroll
    for (int kc = 0; kc < 2; ++kc) {
      const int k0 = kc * 32 + half * 16;
      v4f f0 = *(const v4f*)(xr + k0);
      v4f f1 = *(const v4f*)(xr + k0 + 4);
      v4f f2 = *(const v4f*)(xr + k0 + 8);
      v4f f3 = *(const v4f*)(xr + k0 + 12);
      v16h tf;
#pragma unroll
      for (int e = 0; e < 4; ++e) {
        tf[e]      = (_Float16)f0[e];
        tf[e + 4]  = (_Float16)f1[e];
        tf[e + 8]  = (_Float16)f2[e];
        tf[e + 12] = (_Float16)f3[e];
        sumsq += f0[e] * f0[e] + f1[e] * f1[e] + f2[e] * f2[e] + f3[e] * f3[e];
      }
      a[kc] = tf;
    }
    // Row total = this lane's 32 elements + partner lane (lane^16) 32 elements.
    sumsq += __builtin_bit_cast(
        float, __builtin_amdgcn_ds_swizzle(__builtin_bit_cast(int, sumsq),
                                           0x401F /* SWAPX16 */));
    const float rinv = 1.0f / sumsq;
    // C/D row v+8*half needs rinv held by lane (v+8*half): broadcast via bpermute.
    float rn[8];
#pragma unroll
    for (int v = 0; v < 8; ++v) {
      rn[v] = __builtin_bit_cast(
          float, __builtin_amdgcn_ds_bpermute((v + 8 * half) << 2,
                                              __builtin_bit_cast(int, rinv)));
    }

#pragma unroll
    for (int jt = 0; jt < 4; ++jt) {
      v8f cr = {};
      v8f ci = {};
      cr = __builtin_amdgcn_wmma_f32_16x16x32_f16(false, a[0], false, bre[jt][0],
                                                  (short)0, cr, false, false);
      cr = __builtin_amdgcn_wmma_f32_16x16x32_f16(false, a[1], false, bre[jt][1],
                                                  (short)0, cr, false, false);
      ci = __builtin_amdgcn_wmma_f32_16x16x32_f16(false, a[0], false, bim[jt][0],
                                                  (short)0, ci, false, false);
      ci = __builtin_amdgcn_wmma_f32_16x16x32_f16(false, a[1], false, bim[jt][1],
                                                  (short)0, ci, false, false);
      // D layout: VGPR v -> batch row v + 8*half, lane -> col n.
      float* ob = out + b0 * QDIM + jt * 16 + n;
#pragma unroll
      for (int v = 0; v < 8; ++v) {
        const int row = v + 8 * half;
        ob[(size_t)row * QDIM] = (cr[v] * cr[v] + ci[v] * ci[v]) * rn[v];
      }
    }
  }
}

extern "C" void kernel_launch(void* const* d_in, const int* in_sizes, int n_in,
                              void* d_out, int out_size, void* d_ws, size_t ws_size,
                              hipStream_t stream) {
  const float* x    = (const float*)d_in[0];   // [B, 64] f32
  const float* wrot = (const float*)d_in[1];   // [6, 6] f32
  float* out = (float*)d_out;                  // [B, 64] f32
  _Float16* M = (_Float16*)d_ws;               // 8192 halves = 16 KB scratch

  build_M_kernel<<<1, 64, 0, stream>>>(wrot, M);

  const int B = in_sizes[0] / QDIM;            // 65536
  const int numTiles = B / 16;                 // 4096 16-row tiles
  qenc_wmma_kernel<<<256, 256, 0, stream>>>(x, M, M + 4096, out, numTiles);
}
